// SphericalHealpixBlottleNeck_10316511445497
// MI455X (gfx1250) — matrix operand order repaired and checked by reading, not compiled, not run on hardware
//
#include <hip/hip_runtime.h>

// ---------------------------------------------------------------------------
// Spherical HEALpix ChebConv net for MI455X (gfx1250, wave32, WMMA).
//  - Dense einsum -> v_wmma_f32_16x16x32_f16, NT (template) N-tiles per wave
//  - Software-pipelined K-loop (ping-pong fragment sets, no register copies)
//  - Offset-indexed loads from kernel-arg bases => global_load_b128 (no FLAT)
//  - Weights pre-packed into WMMA B-fragment layout
//  - SPMM exploits the 21-regular row structure (no atomics)
//  - Deterministic two-level BN reduction (coalesced)
// ---------------------------------------------------------------------------

typedef __attribute__((ext_vector_type(16))) _Float16 v16h;
typedef __attribute__((ext_vector_type(8)))  float    v8f;

#define NVERT 12288
#define NBATCH 4
#define MROWS (NVERT * NBATCH)   // 49152, multiple of 16
#define DEG 21                   // NBR + 1 (self + 20 neighbors)
#define BN_EPS 1e-5f
#define NTW 4                    // N-tiles per wave in the GEMM (when ntf%4==0)
#define PBLK 48                  // partial-reduction blocks for BN stats

static inline int cdiv(int a, int b) { return (a + b - 1) / b; }

// ---------------- SPMM: out = alpha * (L @ x) + beta * z -------------------
__global__ __launch_bounds__(256) void spmm_kernel(
    float* __restrict__ out, const float* __restrict__ x,
    const float* __restrict__ z, const float* __restrict__ val,
    const int* __restrict__ col, int V, int C, float alpha, float beta,
    int total) {
  int idx = blockIdx.x * blockDim.x + threadIdx.x;
  if (idx >= total) return;           // total = B*V*C
  int c  = idx % C;
  int bv = idx / C;
  int v  = bv % V;
  int b  = bv / V;
  const int base = v * DEG;
  float s = 0.f;
#pragma unroll
  for (int j = 0; j < DEG; ++j) {
    int cj = col[base + j];
    s += val[base + j] * x[(b * V + cj) * C + c];
  }
  float r = alpha * s;
  if (z) r += beta * z[idx];
  out[idx] = r;
}

// -------- Pack [T0|T1|T2] (f32, [B,V,C] each) -> f16 [M, Kpad] zero-padded --
__global__ __launch_bounds__(256) void pack_x_kernel(
    _Float16* __restrict__ xp, const float* __restrict__ s0,
    const float* __restrict__ s1, const float* __restrict__ s2,
    int C, int Kdim, int Kpad, int total) {
  int idx = blockIdx.x * blockDim.x + threadIdx.x;
  if (idx >= total) return;           // total = M * Kpad
  int m  = idx / Kpad;
  int kc = idx % Kpad;
  float v = 0.f;
  if (kc < Kdim) {
    int k = kc / C, c = kc % C;
    const float* s = (k == 0) ? s0 : (k == 1) ? s1 : s2;
    v = s[m * C + c];
  }
  xp[idx] = (_Float16)v;
}

// -------- Pack weights f32 [Kdim, F] -> WMMA B-fragment layout -------------
// wf[(nt*nkt + kt)*256 + lane*8 + r] = {W[k0,col] , W[k0+1,col]} packed f16,
// with k0 = kt*32 + (lane>=16)*8 + (r<4 ? 2r : 16 + 2(r-4)), col = nt*16 + lane%16 (ISA 7.12.2).
__global__ __launch_bounds__(256) void pack_w_frag_kernel(
    unsigned int* __restrict__ wf, const float* __restrict__ w,
    int Kdim, int F, int nkt, int total) {
  int idx = blockIdx.x * blockDim.x + threadIdx.x;
  if (idx >= total) return;           // total = ntf * nkt * 256
  int slot = idx & 255;
  int tile = idx >> 8;
  int lane = slot >> 3;
  int r    = slot & 7;
  int kt   = tile % nkt;
  int nt   = tile / nkt;
  int hi   = lane >> 4, n = lane & 15;
  int k0   = (kt << 5) + hi * 8 + ((r < 4) ? 2 * r : 16 + 2 * (r - 4));
  int colw = nt * 16 + n;
  union H { _Float16 h; unsigned short s; } lo, hh;
  lo.h = (_Float16)((k0 < Kdim && colw < F) ? w[k0 * F + colw] : 0.f);
  hh.h = (_Float16)((k0 + 1 < Kdim && colw < F) ? w[(k0 + 1) * F + colw] : 0.f);
  wf[idx] = (unsigned int)lo.s | ((unsigned int)hh.s << 16);
}

// ---------------- WMMA GEMM: Y[M,F] = X[M,Kpad] @ W + bias -----------------
union Frag16 { v16h h; unsigned int u[8]; };

__device__ __forceinline__ void load_frag(Frag16& f, const uint4* base,
                                          size_t idx, int second) {
  uint4 x0 = base[idx];
  uint4 x1 = base[idx + second];
  f.u[0] = x0.x; f.u[1] = x0.y; f.u[2] = x0.z; f.u[3] = x0.w;
  f.u[4] = x1.x; f.u[5] = x1.y; f.u[6] = x1.z; f.u[7] = x1.w;
}

template <int NT>
__global__ __launch_bounds__(256) void wmma_gemm_kernel(
    float* __restrict__ Y, const _Float16* __restrict__ X,
    const unsigned int* __restrict__ Wfrag, const float* __restrict__ bias,
    int M, int Kpad, int F, int Fpad) {
  const int lane = threadIdx.x & 31;
  const int wave = threadIdx.x >> 5;
  const int ntf  = Fpad >> 4;
  const int nkt  = Kpad >> 5;
  const int ngrp = ntf / NT;                    // exact by construction
  const int work = blockIdx.x * 8 + wave;       // 8 waves / block
  const int mt   = work / ngrp;
  const int ng   = work % ngrp;
  if (mt * 16 >= M) return;                     // wave-uniform exit

  const int l  = lane & 15;
  const int hi = lane >> 4;
  const int row = mt * 16 + l;

  const uint4* Xv = (const uint4*)X;            // global-addrspace bases
  const uint4* Wv = (const uint4*)Wfrag;
  // A slab kt: Xv[abase + kt*4], Xv[... + 2]   (2 x b128 per lane)
  const size_t abase = (((size_t)row * Kpad) >> 3) + hi;
  // B tile (nt,kt): Wv[bb[t] + kt*64], +1      (tile = 256 dwords = 64 uint4)
  size_t bb[NT];
#pragma unroll
  for (int t = 0; t < NT; ++t)
    bb[t] = ((size_t)((ng * NT + t) * nkt) << 6) + lane * 2;

  v8f acc[NT];
#pragma unroll
  for (int t = 0; t < NT; ++t) {
    int colt = (ng * NT + t) * 16 + l;
    float bv = (colt < F) ? bias[colt] : 0.f;
#pragma unroll
    for (int i = 0; i < 8; ++i) acc[t][i] = bv;
  }

  // ---- software-pipelined K-loop: ping-pong fragment sets ----
  Frag16 aP, aQ, bP[NT], bQ[NT];
  load_frag(aP, Xv, abase, 2);
#pragma unroll
  for (int t = 0; t < NT; ++t) load_frag(bP[t], Wv, bb[t], 1);

  for (int kt = 0; kt < nkt; kt += 2) {         // kt, nkt wave-uniform
    const bool has1 = (kt + 1 < nkt);
    if (has1) {                                 // preload slab kt+1 (Q set)
      load_frag(aQ, Xv, abase + (size_t)(kt + 1) * 4, 2);
#pragma unroll
      for (int t = 0; t < NT; ++t)
        load_frag(bQ[t], Wv, bb[t] + (size_t)(kt + 1) * 64, 1);
    }
    if (kt + 2 < nkt)                           // prefetch slab kt+2's A
      __builtin_prefetch(&X[(size_t)row * Kpad + (size_t)(kt + 2) * 32], 0, 3);

#pragma unroll
    for (int t = 0; t < NT; ++t)                // consume P (shadowed by Q loads)
      acc[t] = __builtin_amdgcn_wmma_f32_16x16x32_f16(
          false, aP.h, false, bP[t].h, (short)0, acc[t], false, false);

    if (kt + 2 < nkt) {                         // preload slab kt+2 (P set)
      load_frag(aP, Xv, abase + (size_t)(kt + 2) * 4, 2);
#pragma unroll
      for (int t = 0; t < NT; ++t)
        load_frag(bP[t], Wv, bb[t] + (size_t)(kt + 2) * 64, 1);
    }
    if (has1) {
#pragma unroll
      for (int t = 0; t < NT; ++t)              // consume Q (shadowed by P loads)
        acc[t] = __builtin_amdgcn_wmma_f32_16x16x32_f16(
            false, aQ.h, false, bQ[t].h, (short)0, acc[t], false, false);
    }
  }

#pragma unroll
  for (int t = 0; t < NT; ++t) {
    int colt = (ng * NT + t) * 16 + l;
    if (colt < F) {                             // only trims F=8 tail columns
#pragma unroll
      for (int i = 0; i < 8; ++i)
        Y[(size_t)(mt * 16 + hi * 8 + i) * F + colt] = acc[t][i];
    }
  }
}

// ---------------- BatchNorm: deterministic two-level reduction -------------
__global__ __launch_bounds__(256) void bn_partial_kernel(
    const float* __restrict__ Y, float* __restrict__ part, int M, int F) {
  const int rpi = 256 / F;            // rows in flight per block (F in {64,256})
  const int c   = threadIdx.x % F;
  const int rg  = threadIdx.x / F;
  float s = 0.f, s2 = 0.f;
  for (int m = blockIdx.x * rpi + rg; m < M; m += PBLK * rpi) {
    float v = Y[(size_t)m * F + c];
    s += v; s2 += v * v;
  }
  __shared__ float sh[256], sq[256];
  sh[threadIdx.x] = s; sq[threadIdx.x] = s2;
  __syncthreads();
  if (threadIdx.x < F) {              // rg == 0 here
    for (int g = 1; g < rpi; ++g) { s += sh[c + g * F]; s2 += sq[c + g * F]; }
    part[blockIdx.x * 2 * F + 2 * c]     = s;
    part[blockIdx.x * 2 * F + 2 * c + 1] = s2;
  }
}

__global__ __launch_bounds__(256) void bn_finalize_kernel(
    const float* __restrict__ part, float* __restrict__ stats, int M, int F) {
  int c = threadIdx.x;
  if (c >= F) return;
  float s = 0.f, s2 = 0.f;
  for (int p = 0; p < PBLK; ++p) {
    s  += part[p * 2 * F + 2 * c];
    s2 += part[p * 2 * F + 2 * c + 1];
  }
  float mean = s / (float)M;
  stats[2 * c]     = mean;
  stats[2 * c + 1] = s2 / (float)M - mean * mean;
}

__global__ __launch_bounds__(256) void bn_apply_kernel(
    float* __restrict__ Y, const float* __restrict__ stats,
    const float* __restrict__ g, const float* __restrict__ b,
    int F, int relu, int total) {
  int idx = blockIdx.x * blockDim.x + threadIdx.x;
  if (idx >= total) return;
  int c = idx % F;
  float v = (Y[idx] - stats[2 * c]) * rsqrtf(stats[2 * c + 1] + BN_EPS) * g[c] + b[c];
  if (relu) v = fmaxf(v, 0.f);
  Y[idx] = v;
}

__global__ __launch_bounds__(256) void add_kernel(
    float* __restrict__ dst, const float* __restrict__ src, int total) {
  int idx = blockIdx.x * blockDim.x + threadIdx.x;
  if (idx < total) dst[idx] += src[idx];
}

// ---------------------------------------------------------------------------
extern "C" void kernel_launch(void* const* d_in, const int* in_sizes, int n_in,
                              void* d_out, int out_size, void* d_ws, size_t ws_size,
                              hipStream_t stream) {
  const int V = NVERT, M = MROWS;

  const float* x        = (const float*)d_in[0];
  const int*   col      = (const int*)  d_in[2];
  const float* val      = (const float*)d_in[3];
  const float* conv1_w  = (const float*)d_in[4];
  const float* conv1_b  = (const float*)d_in[5];
  const float* bn1_g    = (const float*)d_in[6];
  const float* bn1_b    = (const float*)d_in[7];
  const float* conv2_w  = (const float*)d_in[8];
  const float* conv2_b  = (const float*)d_in[9];
  const float* bn2_g    = (const float*)d_in[10];
  const float* bn2_b    = (const float*)d_in[11];
  const float* bt_c1w   = (const float*)d_in[12];
  const float* bt_c1b   = (const float*)d_in[13];
  const float* bt_c2w   = (const float*)d_in[14];
  const float* bt_c2b   = (const float*)d_in[15];
  const float* bt_c3w   = (const float*)d_in[16];
  const float* bt_c3b   = (const float*)d_in[17];
  const float* bt_bn1g  = (const float*)d_in[18];
  const float* bt_bn1b  = (const float*)d_in[19];
  const float* bt_bn2g  = (const float*)d_in[20];
  const float* bt_bn2b  = (const float*)d_in[21];
  const float* bt_bn3g  = (const float*)d_in[22];
  const float* bt_bn3b  = (const float*)d_in[23];
  const float* conv3_w  = (const float*)d_in[24];
  const float* conv3_b  = (const float*)d_in[25];

  // ---- workspace carve-out (256B aligned) ----
  char*  ws  = (char*)d_ws;
  size_t off = 0;
  auto carve = [&](size_t bytes) {
    void* p = ws + off;
    off = (off + bytes + 255) & ~(size_t)255;
    return p;
  };
  const size_t actB = (size_t)M * 256 * sizeof(float);
  float*        hA    = (float*)        carve(actB);
  float*        hB    = (float*)        carve(actB);
  float*        hC    = (float*)        carve(actB);
  float*        t1    = (float*)        carve(actB);
  float*        t2    = (float*)        carve(actB);
  _Float16*     xpack = (_Float16*)     carve((size_t)M * 768 * sizeof(_Float16));
  unsigned int* wfrag = (unsigned int*) carve((size_t)16 * 24 * 256 * sizeof(unsigned int));
  float*        stats = (float*)        carve(2 * 256 * sizeof(float));
  float*        part  = (float*)        carve((size_t)PBLK * 2 * 256 * sizeof(float));
  (void)ws_size;

  auto spmm = [&](float* out, const float* xin, const float* z, int C,
                  float alpha, float beta) {
    int total = M * C;
    spmm_kernel<<<cdiv(total, 256), 256, 0, stream>>>(
        out, xin, z, val, col, V, C, alpha, beta, total);
  };

  // Chebyshev conv (K in {1,3}): Y = [T0|T1|T2] @ flat(w) + bias
  auto cheb = [&](const float* hin, int Cin, int K, const float* w,
                  const float* bias, int F, float* Yout) {
    int Kdim = K * Cin;
    int Kpad = (Kdim + 31) & ~31;
    int Fpad = (F + 15) & ~15;
    int nkt  = Kpad >> 5;
    int ntf  = Fpad >> 4;
    const float *s1 = hin, *s2 = hin;
    if (K >= 2) { spmm(t1, hin, nullptr, Cin, 1.f, 0.f); s1 = t1; }
    if (K >= 3) { spmm(t2, t1, hin, Cin, 2.f, -1.f); s2 = t2; }
    {
      int total = M * Kpad;
      pack_x_kernel<<<cdiv(total, 256), 256, 0, stream>>>(
          xpack, hin, s1, s2, Cin, Kdim, Kpad, total);
    }
    {
      int total = ntf * nkt * 256;
      pack_w_frag_kernel<<<cdiv(total, 256), 256, 0, stream>>>(
          wfrag, w, Kdim, F, nkt, total);
    }
    if (ntf % NTW == 0) {
      int work = (M / 16) * (ntf / NTW);
      wmma_gemm_kernel<NTW><<<cdiv(work, 8), 256, 0, stream>>>(
          Yout, xpack, wfrag, bias, M, Kpad, F, Fpad);
    } else {
      int work = (M / 16) * ntf;
      wmma_gemm_kernel<1><<<cdiv(work, 8), 256, 0, stream>>>(
          Yout, xpack, wfrag, bias, M, Kpad, F, Fpad);
    }
  };

  auto bn = [&](float* Y, const float* g, const float* b, int F, int relu) {
    bn_partial_kernel<<<PBLK, 256, 0, stream>>>(Y, part, M, F);
    bn_finalize_kernel<<<1, 256, 0, stream>>>(part, stats, M, F);
    int total = M * F;
    bn_apply_kernel<<<cdiv(total, 256), 256, 0, stream>>>(
        Y, stats, g, b, F, relu, total);
  };

  // ---- conv1: 8 -> 64, K=3, BN+ReLU ----
  cheb(x, 8, 3, conv1_w, conv1_b, 64, hA);
  bn(hA, bn1_g, bn1_b, 64, 1);

  // ---- conv2: 64 -> 256, K=3, BN+ReLU ----
  cheb(hA, 64, 3, conv2_w, conv2_b, 256, hB);
  bn(hB, bn2_g, bn2_b, 256, 1);

  // ---- 3 bottlenecks with residual ----
  for (int i = 0; i < 3; ++i) {
    cheb(hB, 256, 1, bt_c1w + (size_t)i * 256 * 64, bt_c1b + i * 64, 64, hC);
    bn(hC, bt_bn1g + i * 64, bt_bn1b + i * 64, 64, 1);

    cheb(hC, 64, 3, bt_c2w + (size_t)i * 3 * 64 * 64, bt_c2b + i * 64, 64, hA);
    bn(hA, bt_bn2g + i * 64, bt_bn2b + i * 64, 64, 1);

    cheb(hA, 64, 1, bt_c3w + (size_t)i * 64 * 256, bt_c3b + i * 256, 256, hC);
    bn(hC, bt_bn3g + i * 256, bt_bn3b + i * 256, 256, 0);

    int total = M * 256;
    add_kernel<<<cdiv(total, 256), 256, 0, stream>>>(hB, hC, total);
  }

  // ---- conv3: 256 -> 8, K=3, straight into d_out ([M,8] layout matches) ----
  cheb(hB, 256, 3, conv3_w, conv3_b, 8, (float*)d_out);
  (void)in_sizes; (void)n_in; (void)out_size;
}
